// GATLayer_65146063946156
// MI455X (gfx1250) — matrix operand order repaired and checked by reading, not compiled, not run on hardware
//
#include <hip/hip_runtime.h>
#include <hip/hip_bf16.h>

// Problem constants (reference: B,N,D,H,O = 2,4096,128,4,32)
#define NB 2
#define NN 4096
#define ND 128
#define NH 4
#define NO 32
#define SCALE 0.17677669529663687f   // 1/sqrt(32)

typedef _Float16 v8h  __attribute__((ext_vector_type(8)));
typedef _Float16 v16h __attribute__((ext_vector_type(16)));
typedef float    v8f  __attribute__((ext_vector_type(8)));

__device__ __forceinline__ v16h cat16(v8h a, v8h b) {
    return __builtin_shufflevector(a, b, 0,1,2,3,4,5,6,7,8,9,10,11,12,13,14,15);
}

__device__ __forceinline__ v8f wmma_f16(v16h a, v16h b, v8f c) {
    // D = A(16x32 f16) * B(32x16 f16) + C(16x16 f32)
    return __builtin_amdgcn_wmma_f32_16x16x32_f16(false, a, false, b, (short)0, c, false, false);
}

// ---------------------------------------------------------------------------
// Kernel 1: q[b,h,n,o] = sum_d x[b,n,d] * W[d,h,o]; store f16 row-major and
// transposed ([b,h,o,n]) so the aggregation-WMMA B-fragments load contiguous K.
// ---------------------------------------------------------------------------
__global__ void k_proj(const float* __restrict__ x, const float* __restrict__ W,
                       _Float16* __restrict__ q, _Float16* __restrict__ qt) {
    __shared__ float xs[ND];
    int bn = blockIdx.x;              // b*N + n
    int b  = bn >> 12;
    int n  = bn & (NN - 1);
    int t  = threadIdx.x;             // 0..127  ->  (h,o)
    xs[t] = x[(size_t)bn * ND + t];
    __syncthreads();
    int h = t >> 5, o = t & 31;
    float acc = 0.f;
    #pragma unroll 8
    for (int d = 0; d < ND; ++d)
        acc = fmaf(xs[d], W[d * (NH * NO) + t], acc);   // W[d][h][o], t = h*32+o
    _Float16 v = (_Float16)acc;
    q [(((size_t)b * NH + h) * NN + n) * NO + o] = v;
    qt[(((size_t)b * NH + h) * NO + o) * NN + n] = v;
}

// ---------------------------------------------------------------------------
// Kernel 2: softmax denominators. One wave = 16 query rows of one (b,h).
// Scores via v_wmma_f32_16x16x32_f16; exp accumulated per lane; 16-lane
// butterfly reduction (C-tile rows are striped across 16-lane halves).
// Scores are O(10) max -> exp cannot overflow f32; skip max-subtraction.
// ---------------------------------------------------------------------------
__global__ void k_rowsum(const _Float16* __restrict__ q, const float* __restrict__ adj,
                         float* __restrict__ sums) {
    int blk    = blockIdx.x;          // (b*H+h)*32 + rowblk
    int bh     = blk >> 5;
    int rowblk = blk & 31;
    int wave = threadIdx.x >> 5;
    int lane = threadIdx.x & 31;
    int m    = lane & 15;
    int half = lane >> 4;
    int row0 = rowblk * 128 + wave * 16;

    const _Float16* qh = q + (size_t)bh * NN * NO;

    // A fragment: rows row0..row0+15, K = O-dim (32)
    const _Float16* ar = qh + (size_t)(row0 + m) * NO + half * 8;
    v16h afrag = cat16(*(const v8h*)ar, *(const v8h*)(ar + 16));

    float acc[8];
    #pragma unroll
    for (int r = 0; r < 8; ++r) acc[r] = 0.f;

    for (int j = 0; j < NN; j += 16) {
        const _Float16* br = qh + (size_t)(j + m) * NO + half * 8;
        v16h bfrag = cat16(*(const v8h*)br, *(const v8h*)(br + 16));
        v8f c = {};
        c = wmma_f16(afrag, bfrag, c);
        #pragma unroll
        for (int r = 0; r < 8; ++r) {
            int rr = row0 + r + 8 * half;
            float a = adj[(size_t)rr * NN + j + m];
            acc[r] += __expf(c[r] * SCALE + a);
        }
    }
    #pragma unroll
    for (int r = 0; r < 8; ++r) {
        float s = acc[r];
        s += __shfl_xor(s, 1, 32);
        s += __shfl_xor(s, 2, 32);
        s += __shfl_xor(s, 4, 32);
        s += __shfl_xor(s, 8, 32);
        acc[r] = s;
    }
    if (m == 0) {
        #pragma unroll
        for (int r = 0; r < 8; ++r)
            sums[(size_t)bh * NN + row0 + r + 8 * half] = acc[r];
    }
}

// ---------------------------------------------------------------------------
// Kernel 3: one wave = 16 query rows, all 4 heads. Recompute scores (WMMA),
// normalize, accumulate head-mean tile, stage w as f16 in per-wave LDS,
// reload in A-fragment layout, WMMA against q^T to accumulate output.
// w_mean written coalesced once; attention matrix never hits HBM.
// ---------------------------------------------------------------------------
__global__ void k_attn(const _Float16* __restrict__ q, const _Float16* __restrict__ qt,
                       const float* __restrict__ adj, const float* __restrict__ sums,
                       float* __restrict__ out, float* __restrict__ wmean) {
    int b      = blockIdx.x >> 5;     // 32 row-blocks of 128 rows
    int rowblk = blockIdx.x & 31;
    int wave = threadIdx.x >> 5;
    int lane = threadIdx.x & 31;
    int m    = lane & 15;
    int half = lane >> 4;
    int row0 = rowblk * 128 + wave * 16;

    __shared__ _Float16 wstage[8][16 * 32];   // per-wave 16x32 f16 staging
    _Float16* ws = wstage[wave];

    v16h aq[NH];
    float inv[NH][8];
    #pragma unroll
    for (int h = 0; h < NH; ++h) {
        const _Float16* qh = q + (size_t)(b * NH + h) * NN * NO;
        const _Float16* ar = qh + (size_t)(row0 + m) * NO + half * 8;
        aq[h] = cat16(*(const v8h*)ar, *(const v8h*)(ar + 16));
        #pragma unroll
        for (int r = 0; r < 8; ++r)
            inv[h][r] = 1.0f / sums[(size_t)(b * NH + h) * NN + row0 + r + 8 * half];
    }

    v8f oc[NH][2];
    #pragma unroll
    for (int h = 0; h < NH; ++h)
        #pragma unroll
        for (int of = 0; of < 2; ++of)
            oc[h][of] = (v8f){};

    for (int j = 0; j < NN; j += 32) {
        float wm[16];
        #pragma unroll
        for (int i = 0; i < 16; ++i) wm[i] = 0.f;

        #pragma unroll
        for (int h = 0; h < NH; ++h) {
            const _Float16* qh  = q  + (size_t)(b * NH + h) * NN * NO;
            const _Float16* qth = qt + (size_t)(b * NH + h) * NO * NN;

            #pragma unroll
            for (int j2 = 0; j2 < 2; ++j2) {
                int jc = j + j2 * 16;
                const _Float16* br = qh + (size_t)(jc + m) * NO + half * 8;
                v16h bq = cat16(*(const v8h*)br, *(const v8h*)(br + 16));
                v8f c = {};
                c = wmma_f16(aq[h], bq, c);
                #pragma unroll
                for (int r = 0; r < 8; ++r) {
                    float a = adj[(size_t)(row0 + r + 8 * half) * NN + jc + m];
                    float w = __expf(c[r] * SCALE + a) * inv[h][r];
                    wm[j2 * 8 + r] += 0.25f * w;
                    ws[(r + 8 * half) * 32 + j2 * 16 + m] = (_Float16)w;
                }
            }
            // reload w tile as A fragment (16x32 f16) from LDS
            const _Float16* wr = ws + m * 32 + half * 8;
            v16h wa = cat16(*(const v8h*)wr, *(const v8h*)(wr + 16));
            // B fragments from q^T: K = column index (j..j+31), N = O columns
            #pragma unroll
            for (int of = 0; of < 2; ++of) {
                const _Float16* btr = qth + (size_t)(of * 16 + m) * NN + j + half * 8;
                v16h bo = cat16(*(const v8h*)btr, *(const v8h*)(btr + 16));
                oc[h][of] = wmma_f16(wa, bo, oc[h][of]);
            }
        }
        // head-mean tile -> global (written exactly once, coalesced per row-half)
        #pragma unroll
        for (int j2 = 0; j2 < 2; ++j2)
            #pragma unroll
            for (int r = 0; r < 8; ++r)
                wmean[(size_t)b * NN * NN + (size_t)(row0 + r + 8 * half) * NN + j + j2 * 16 + m]
                    = wm[j2 * 8 + r];
    }

    // out[b][n][h*32 + o]
    #pragma unroll
    for (int h = 0; h < NH; ++h)
        #pragma unroll
        for (int of = 0; of < 2; ++of)
            #pragma unroll
            for (int r = 0; r < 8; ++r)
                out[((size_t)b * NN + row0 + r + 8 * half) * (NH * NO) + h * NO + of * 16 + m]
                    = oc[h][of][r];
}

// ---------------------------------------------------------------------------
extern "C" void kernel_launch(void* const* d_in, const int* in_sizes, int n_in,
                              void* d_out, int out_size, void* d_ws, size_t ws_size,
                              hipStream_t stream) {
    const float* x   = (const float*)d_in[0];   // [B,N,D]
    const float* adj = (const float*)d_in[1];   // [N,N]
    const float* W   = (const float*)d_in[2];   // [D,H,O]

    float* out   = (float*)d_out;                                   // [B,N,H*O]
    float* wmean = out + (size_t)NB * NN * NH * NO;                 // [B,N,N]

    char* ws = (char*)d_ws;
    const size_t qbytes = (size_t)NB * NH * NN * NO * sizeof(_Float16); // 2 MB
    _Float16* q    = (_Float16*)ws;
    _Float16* qt   = (_Float16*)(ws + qbytes);
    float*    sums = (float*)(ws + 2 * qbytes);                     // B*H*N f32

    k_proj  <<<NB * NN,                 128, 0, stream>>>(x, W, q, qt);
    k_rowsum<<<NB * NH * (NN / 128),    256, 0, stream>>>(q, adj, sums);
    k_attn  <<<NB * (NN / 128),         256, 0, stream>>>(q, qt, adj, sums, out, wmean);
}
